// EnhancedMessagePassing_47974784696386
// MI455X (gfx1250) — matrix operand order repaired and checked by reading, not compiled, not run on hardware
//
#include <hip/hip_runtime.h>

// ---------------------------------------------------------------------------
// EnhancedMessagePassing for MI455X (gfx1250, wave32, WMMA f32_16x16x32_f16)
//
//   P0 prep:    fragment-pack W1 (f16) and BigB [4160,64] (f16)
//   P1 hmid:    relu(EF @ W1 + b1) -> f16 [E,64]                 (WMMA)
//   P2 message: GEMM [E,4160]x[4160,64], A = hmid (x) src on the fly,
//               B staged chunk-wise into double-buffered LDS with
//               GLOBAL_LOAD_ASYNC_TO_LDS_B128 (ASYNCcnt), shared by 8 waves;
//               each wave computes 32 edges; epilogue scatters into segment
//               sums with atomics                                (WMMA)
//   P3 counts / mean
//   P4 gx = agg @ Wx + bx (fp32 VALU, keeps GRU input precise)
//   P5 GRU: single persistent workgroup, Wh in LDS, sequential scan
// ---------------------------------------------------------------------------

typedef __attribute__((ext_vector_type(16))) _Float16 v16h;
typedef __attribute__((ext_vector_type(8)))  _Float16 v8h;
typedef __attribute__((ext_vector_type(8)))  float    v8f;

union AFrag { v16h v; v8h h[2]; };

#define UDIM    64
#define KTOT    4160            // 64*64 (hmid (x) src) + 64 (src tail for b2)
#define KSTEPS  130             // KTOT / 32
#define NTILES  4               // 64 / 16
#define CK      13              // k-steps per LDS chunk
#define NCHUNK  10              // KSTEPS / CK
#define CHUNK_ELEMS (CK * 2048) // f16 per chunk
#define CHUNK_BYTES (CHUNK_ELEMS * 2)   // 53248
#define TSTRIDE 72              // padded f16 row stride for LDS tiles
#define MSG_LDS (2*CHUNK_BYTES + 8*2*32*TSTRIDE*2 + 8*32*4)

// ---- P0: fragment-pack weights -------------------------------------------
__global__ void prep_kernel(const float* __restrict__ W1,
                            const float* __restrict__ W2,
                            const float* __restrict__ b2,
                            _Float16* __restrict__ W1pack,
                            _Float16* __restrict__ Bpack) {
  int idx = blockIdx.x * blockDim.x + threadIdx.x;
  int stride = gridDim.x * blockDim.x;
  const int total = KSTEPS * NTILES * 32 * 16;
  for (int i = idx; i < total; i += stride) {
    int e    = i & 15;
    int lane = (i >> 4) & 31;
    int nt   = (i >> 9) & 3;
    int ks   = i >> 11;
    int K    = ks * 32 + ((e >> 3) << 4) + ((lane >> 4) << 3) + (e & 7);   // B frag: K = (e>>3)*16 + half*8 + (e&7)
    int n    = nt * 16 + (lane & 15);
    float v;
    if (K < 4096) {
      int kh = K >> 6, j = K & 63;
      v = W2[kh * 4096 + n * 64 + j];
    } else {
      v = b2[n * 64 + (K - 4096)];
    }
    Bpack[i] = (_Float16)v;
  }
  for (int i = idx; i < 2 * NTILES * 32 * 16; i += stride) {  // 4096
    int e    = i & 15;
    int lane = (i >> 4) & 31;
    int nt   = (i >> 9) & 3;
    int s    = i >> 11;
    int K    = s * 32 + ((e >> 3) << 4) + ((lane >> 4) << 3) + (e & 7);
    int n    = nt * 16 + (lane & 15);
    W1pack[i] = (_Float16)W1[K * 64 + n];
  }
}

// ---- P1: hmid = relu(EF @ W1 + b1) ---------------------------------------
__global__ __launch_bounds__(32)
void hmid_kernel(const float* __restrict__ EF,
                 const _Float16* __restrict__ W1pack,
                 const float* __restrict__ b1,
                 _Float16* __restrict__ hmid) {
  const int tile = blockIdx.x;
  const int lane = threadIdx.x & 31;
  const int half = lane >> 4;
  const int m    = lane & 15;
  const int row  = tile * 16 + m;

  AFrag a[2];
  #pragma unroll
  for (int s = 0; s < 2; ++s) {
    #pragma unroll
    for (int run = 0; run < 2; ++run) {
      int start = s * 32 + run * 16 + half * 8;
      float4 f0 = *(const float4*)(EF + (size_t)row * 64 + start);
      float4 f1 = *(const float4*)(EF + (size_t)row * 64 + start + 4);
      v8h t;
      t[0] = (_Float16)f0.x; t[1] = (_Float16)f0.y;
      t[2] = (_Float16)f0.z; t[3] = (_Float16)f0.w;
      t[4] = (_Float16)f1.x; t[5] = (_Float16)f1.y;
      t[6] = (_Float16)f1.z; t[7] = (_Float16)f1.w;
      a[s].h[run] = t;
    }
  }

  v8f c[NTILES] = {};
  #pragma unroll
  for (int nt = 0; nt < NTILES; ++nt) {
    #pragma unroll
    for (int s = 0; s < 2; ++s) {
      v16h b = *(const v16h*)(W1pack + ((s * NTILES + nt) * 32 + lane) * 16);
      c[nt] = __builtin_amdgcn_wmma_f32_16x16x32_f16(
          false, a[s].v, false, b, (short)0, c[nt], false, false);
    }
  }

  #pragma unroll
  for (int nt = 0; nt < NTILES; ++nt) {
    #pragma unroll
    for (int v = 0; v < 8; ++v) {
      int M = half * 8 + v;
      int n = nt * 16 + m;
      float val = fmaxf(c[nt][v] + b1[n], 0.0f);
      hmid[(size_t)(tile * 16 + M) * 64 + n] = (_Float16)val;
    }
  }
}

// ---- P2: messages GEMM + segment-sum scatter -----------------------------
__global__ __launch_bounds__(256)
void message_kernel(const _Float16* __restrict__ hmid,
                    const float* __restrict__ NF,
                    const int* __restrict__ eidx,     // [2,E]
                    const _Float16* __restrict__ Bpack,
                    float* __restrict__ sums,
                    int E, int nUnits) {
  extern __shared__ char smem[];
  _Float16* Bbuf  = (_Float16*)smem;                         // 2 chunks
  _Float16* tiles = (_Float16*)(smem + 2 * CHUNK_BYTES);
  int* tgtAll     = (int*)(smem + 2 * CHUNK_BYTES + 8 * 2 * 32 * TSTRIDE * 2);

  const int tid  = threadIdx.x;
  const int lane = tid & 31;
  const int wid  = tid >> 5;
  const int half = lane >> 4;
  const int m    = lane & 15;

  _Float16* hmidS = tiles + wid * (2 * 32 * TSTRIDE);
  _Float16* srcS  = hmidS + 32 * TSTRIDE;
  int* tgtW = tgtAll + wid * 32;

  const int unit    = blockIdx.x * 8 + wid;    // 32-edge unit
  const bool active = unit < nUnits;
  const int base    = unit * 32;

  // LDS byte address of Bbuf (ISA: generic addr low 32 bits = LDS offset)
  const unsigned bbufBase = (unsigned)(unsigned long long)(Bbuf);

  // Async stage one 53248B chunk of packed B into LDS buffer `buf`.
  // 256 threads x 13 x 16B = CHUNK_BYTES. GVS mode: saddr + 32-bit vaddr.
  auto stage_async = [&](int chunk, int buf) {
    const char* srcBase = (const char*)Bpack + (size_t)chunk * CHUNK_BYTES;
    unsigned dst0 = bbufBase + (unsigned)(buf * CHUNK_BYTES) + (unsigned)tid * 16u;
    unsigned src0 = (unsigned)tid * 16u;
    #pragma unroll
    for (int i = 0; i < CK; ++i) {
      unsigned dst = dst0 + (unsigned)i * 4096u;
      unsigned src = src0 + (unsigned)i * 4096u;
      asm volatile("global_load_async_to_lds_b128 %0, %1, %2"
                   :: "v"(dst), "v"(src), "s"(srcBase)
                   : "memory");
    }
  };

  stage_async(0, 0);   // prologue: chunk 0 -> buffer 0 (async, overlaps below)

  if (active) {
    for (int t = lane; t < 32 * 8; t += 32) {
      int r = t >> 3, c8 = (t & 7) * 8;
      *(v8h*)&hmidS[r * TSTRIDE + c8] =
          *(const v8h*)(hmid + (size_t)(base + r) * 64 + c8);
      int node = eidx[base + r];
      float4 f0 = *(const float4*)(NF + (size_t)node * 64 + c8);
      float4 f1 = *(const float4*)(NF + (size_t)node * 64 + c8 + 4);
      v8h sv;
      sv[0] = (_Float16)f0.x; sv[1] = (_Float16)f0.y;
      sv[2] = (_Float16)f0.z; sv[3] = (_Float16)f0.w;
      sv[4] = (_Float16)f1.x; sv[5] = (_Float16)f1.y;
      sv[6] = (_Float16)f1.z; sv[7] = (_Float16)f1.w;
      *(v8h*)&srcS[r * TSTRIDE + c8] = sv;
    }
    tgtW[lane] = eidx[E + base + lane];
  }

  asm volatile("s_wait_asynccnt 0\n\ts_wait_xcnt 0" ::: "memory");
  __syncthreads();

  v8f acc[2][NTILES] = {};
  for (int c = 0; c < NCHUNK; ++c) {
    const int cur = c & 1;
    // kick off async staging of the next chunk; TDM-style overlap with WMMA
    if (c + 1 < NCHUNK) stage_async(c + 1, 1 - cur);

    if (active) {
      for (int ksl = 0; ksl < CK; ++ksl) {
        const int ksg = c * CK + ksl;
        // A[m,K] = hmid[m,K>>6]*src[m,K&63] (K<4096) else src[m,K-4096];
        // each lane's 16 K-values are two aligned 8-runs -> 1 broadcast + vec
        AFrag a[2];
        #pragma unroll
        for (int mt = 0; mt < 2; ++mt) {
          const int rloc = mt * 16 + m;
          #pragma unroll
          for (int run = 0; run < 2; ++run) {
            int start = ksg * 32 + run * 16 + half * 8;  // wave-uniform branch
            v8h av;
            if (start < 4096) {
              _Float16 hm = hmidS[rloc * TSTRIDE + (start >> 6)];
              v8h sv = *(const v8h*)&srcS[rloc * TSTRIDE + (start & 63)];
              #pragma unroll
              for (int u = 0; u < 8; ++u) av[u] = hm * sv[u];
            } else {
              av = *(const v8h*)&srcS[rloc * TSTRIDE + (start - 4096)];
            }
            a[mt].h[run] = av;
          }
        }
        #pragma unroll
        for (int nt = 0; nt < NTILES; ++nt) {
          v16h b = *(const v16h*)&Bbuf[cur * CHUNK_ELEMS +
                                       ((ksl * NTILES + nt) * 32 + lane) * 16];
          acc[0][nt] = __builtin_amdgcn_wmma_f32_16x16x32_f16(
              false, a[0].v, false, b, (short)0, acc[0][nt], false, false);
          acc[1][nt] = __builtin_amdgcn_wmma_f32_16x16x32_f16(
              false, a[1].v, false, b, (short)0, acc[1][nt], false, false);
        }
      }
    }
    // async copies done + address regs reusable, then block-wide visibility
    asm volatile("s_wait_asynccnt 0\n\ts_wait_xcnt 0" ::: "memory");
    __syncthreads();
  }

  if (active) {
    #pragma unroll
    for (int mt = 0; mt < 2; ++mt)
      #pragma unroll
      for (int nt = 0; nt < NTILES; ++nt)
        #pragma unroll
        for (int v = 0; v < 8; ++v) {
          int M = mt * 16 + half * 8 + v;
          atomicAdd(&sums[(size_t)tgtW[M] * 64 + nt * 16 + m], acc[mt][nt][v]);
        }
  }
}

// ---- P3: counts + mean ----------------------------------------------------
__global__ void count_kernel(const int* __restrict__ eidx, float* __restrict__ cnt, int E) {
  int i = blockIdx.x * blockDim.x + threadIdx.x;
  if (i < E) atomicAdd(&cnt[eidx[E + i]], 1.0f);
}

__global__ void agg_kernel(float* __restrict__ sums, const float* __restrict__ cnt, int N) {
  int i = blockIdx.x * blockDim.x + threadIdx.x;
  if (i < N * 64) sums[i] /= fmaxf(cnt[i >> 6], 1.0f);
}

// ---- P4: gx = agg @ Wx + bx (fp32) ---------------------------------------
__global__ void gx_kernel(const float* __restrict__ agg, const float* __restrict__ Wx,
                          const float* __restrict__ bx, float* __restrict__ gx, int N) {
  int i = blockIdx.x * blockDim.x + threadIdx.x;
  if (i >= N * 192) return;
  int ccol = i % 192, n = i / 192;
  float acc = bx[ccol];
  const float* ar = agg + (size_t)n * 64;
  #pragma unroll 8
  for (int k = 0; k < 64; ++k) acc += ar[k] * Wx[k * 192 + ccol];
  gx[i] = acc;
}

// ---- P5: sequential GRU scan, single persistent workgroup ----------------
__global__ __launch_bounds__(192)
void gru_kernel(const float* __restrict__ gx, const float* __restrict__ Wh,
                const float* __restrict__ bh, const float* __restrict__ h0,
                float* __restrict__ out, int N) {
  __shared__ float Wh_s[64][192];   // 48 KB, conflict-free stride-192 reads
  __shared__ float h_s[64];
  __shared__ float gh_s[192];
  const int t = threadIdx.x;
  for (int i = t; i < 64 * 192; i += 192) Wh_s[i / 192][i % 192] = Wh[i];
  if (t < 64) h_s[t] = h0[t];
  const float bhv = bh[t];
  __syncthreads();

  for (int node = 0; node < N; ++node) {
    float acc = bhv;
    #pragma unroll 8
    for (int k = 0; k < 64; ++k) acc += h_s[k] * Wh_s[k][t];
    gh_s[t] = acc;
    __syncthreads();
    float hn = 0.0f;
    if (t < 64) {
      const float* g = gx + (size_t)node * 192;
      float z    = 1.0f / (1.0f + __expf(-(g[t]       + gh_s[t])));
      float r    = 1.0f / (1.0f + __expf(-(g[64 + t]  + gh_s[64 + t])));
      float cand = tanhf(g[128 + t] + r * gh_s[128 + t]);
      hn = z * h_s[t] + (1.0f - z) * cand;
    }
    __syncthreads();
    if (t < 64) h_s[t] = hn;
    __syncthreads();
  }
  if (t < 64) { out[t] = h_s[t]; out[64 + t] = h_s[t]; }
}

// ---------------------------------------------------------------------------
extern "C" void kernel_launch(void* const* d_in, const int* in_sizes, int n_in,
                              void* d_out, int out_size, void* d_ws, size_t ws_size,
                              hipStream_t stream) {
  const float* NF = (const float*)d_in[0];
  const float* EF = (const float*)d_in[1];
  const int*   EI = (const int*)d_in[2];
  const float* H0 = (const float*)d_in[3];
  const float* W1 = (const float*)d_in[4];
  const float* b1 = (const float*)d_in[5];
  const float* W2 = (const float*)d_in[6];
  const float* b2 = (const float*)d_in[7];
  const float* Wx = (const float*)d_in[8];
  const float* Wh = (const float*)d_in[9];
  const float* bx = (const float*)d_in[10];
  const float* bh = (const float*)d_in[11];
  float* out = (float*)d_out;

  const int N = in_sizes[0] / UDIM;   // 10000
  const int E = in_sizes[1] / UDIM;   // 100000 (multiple of 32)

  char* ws = (char*)d_ws;
  size_t off = 0;
  auto alloc = [&](size_t bytes) -> void* {
    off = (off + 255) & ~(size_t)255;
    void* p = ws + off;
    off += bytes;
    return p;
  };
  _Float16* Bpack  = (_Float16*)alloc((size_t)KSTEPS * NTILES * 32 * 16 * sizeof(_Float16));
  _Float16* W1pack = (_Float16*)alloc((size_t)2 * NTILES * 32 * 16 * sizeof(_Float16));
  _Float16* hmid   = (_Float16*)alloc((size_t)E * 64 * sizeof(_Float16));
  float*    sums   = (float*)alloc((size_t)N * 64 * sizeof(float));
  float*    cnt    = (float*)alloc((size_t)N * sizeof(float));
  float*    gx     = (float*)alloc((size_t)N * 192 * sizeof(float));

  hipMemsetAsync(sums, 0, (size_t)N * 64 * sizeof(float), stream);
  hipMemsetAsync(cnt, 0, (size_t)N * sizeof(float), stream);

  prep_kernel<<<(KSTEPS * NTILES * 32 * 16 + 255) / 256, 256, 0, stream>>>(
      W1, W2, b2, W1pack, Bpack);
  hmid_kernel<<<E / 16, 32, 0, stream>>>(EF, W1pack, b1, hmid);

  const int nUnits = E / 32;
  message_kernel<<<(nUnits + 7) / 8, 256, MSG_LDS, stream>>>(
      hmid, NF, EI, Bpack, sums, E, nUnits);

  count_kernel<<<(E + 255) / 256, 256, 0, stream>>>(EI, cnt, E);
  agg_kernel<<<(N * 64 + 255) / 256, 256, 0, stream>>>(sums, cnt, N);
  gx_kernel<<<(N * 192 + 255) / 256, 256, 0, stream>>>(sums, Wx, bx, gx, N);
  gru_kernel<<<1, 192, 0, stream>>>(gx, Wh, bh, H0, out, N);
}
